// LengthRegulator_70068096467969
// MI455X (gfx1250) — compile-verified, matrix operand order checked by reading, not run on hardware
//
#include <hip/hip_runtime.h>

// ---------------------------------------------------------------------------
// LengthRegulator for MI455X (gfx1250, wave32, WMMA)
//   conv1d(K=3,stride1,VALID) -> LN -> ReLU  (x2)  as bf16 WMMA GEMMs
//   duration head fused into conv2 epilogue
//   cumsum + searchsorted gather with nontemporal streaming stores
//
// Roofline: output stream (16x4096x256 fp32 = 67MB) dominates (~3us @23.3TB/s);
// conv GEMMs (2 x ~3.2 GFLOP bf16) are negligible under the WMMA ceiling, so
// they are fused with bias+LayerNorm+ReLU (and the duration head) to avoid
// materializing intermediates in fp32.
// ---------------------------------------------------------------------------

typedef __attribute__((ext_vector_type(16))) __bf16 v16bf;
typedef __attribute__((ext_vector_type(8)))  float  v8f;
typedef __attribute__((ext_vector_type(4)))  float  f4;
typedef __bf16 bf16_t;

#define NB    16
#define LSEQ  512
#define DIM   256
#define TMAX  4096
#define KRED  768   // 3 * 256 (im2col reduction: kk = k*256 + d)

// ---------------- elementwise casts / weight packing -----------------------

__global__ void cast_x_bf16(const float* __restrict__ x, bf16_t* __restrict__ xb, int n) {
    int i = blockIdx.x * blockDim.x + threadIdx.x;
    if (i < n) xb[i] = (bf16_t)x[i];
}

// w: [256][256][3] (O,I,K) -> wb: [o][kk] with kk = k*256 + i  (row-contig in kk)
__global__ void pack_w_bf16(const float* __restrict__ w, bf16_t* __restrict__ wb) {
    int i = blockIdx.x * blockDim.x + threadIdx.x;
    if (i < 256 * KRED) {
        int o  = i / KRED;
        int kk = i - o * KRED;
        int k  = kk >> 8;
        int d  = kk & 255;
        wb[i] = (bf16_t)w[(o * 256 + d) * 3 + k];
    }
}

// ---------------- fused conv(GEMM) + bias + LayerNorm + ReLU ---------------
// Block = 256 threads = 8 waves. Each block owns a 16-row x 256-col tile.
// Wave w owns columns [32w, 32w+32): two f32 16x16 accumulators.
// A rows are contiguous KRED slices of the [n][Lin][256] activation tensor.

__device__ inline v16bf load_frag(const bf16_t* p) {
    union { uint4 u[2]; v16bf v; } f;
    f.u[0] = *(const uint4*)(p);
    f.u[1] = *(const uint4*)(p + 16);
    return f.v;
}

template <bool DUR>
__global__ __launch_bounds__(256, 2)
void conv_ln_wmma(const bf16_t* __restrict__ A,     // [n][Lin][256] bf16
                  const bf16_t* __restrict__ Bw,    // [256][KRED]   bf16
                  const float*  __restrict__ bias,
                  const float*  __restrict__ gamma,
                  const float*  __restrict__ beta,
                  bf16_t*       __restrict__ Hout,  // [M][256] bf16 (if !DUR)
                  const float*  __restrict__ lin_w,
                  const float*  __restrict__ lin_b,
                  float*        __restrict__ dur,   // [M] (if DUR)
                  int Lout, int Lin)
{
    const int tid  = threadIdx.x;
    const int wave = tid >> 5;
    const int lane = tid & 31;
    const int lo16 = lane & 15;
    const int hi   = lane >> 4;

    // A fragment source: lane holds row M = lo16, K-half = hi (ISA 16-bit A layout)
    const int mi = blockIdx.x * 16 + lo16;
    const int n  = mi / Lout;
    const int l  = mi - n * Lout;
    const bf16_t* arow = A + (size_t)(n * Lin + l) * DIM + hi * 8;

    // B fragment source: lane holds column N = lo16 of this wave's two N-tiles
    const int col0 = wave * 32 + lo16;
    const int col1 = col0 + 16;
    const bf16_t* brow0 = Bw + (size_t)col0 * KRED + hi * 8;
    const bf16_t* brow1 = Bw + (size_t)col1 * KRED + hi * 8;

    v8f c0 = {};
    v8f c1 = {};

    for (int kk = 0; kk < KRED; kk += 32) {
        v16bf a  = load_frag(arow  + kk);
        v16bf b0 = load_frag(brow0 + kk);
        v16bf b1 = load_frag(brow1 + kk);
        // near-scope prefetch of the A stream (locality 3 -> WGP scope:
        // fills all cache levels on a miss)
        __builtin_prefetch(arow + kk + 256, 0, 3);   // global_prefetch_b8
        c0 = __builtin_amdgcn_wmma_f32_16x16x32_bf16(false, a, false, b0,
                                                     (short)0, c0, false, false);
        c1 = __builtin_amdgcn_wmma_f32_16x16x32_bf16(false, a, false, b1,
                                                     (short)0, c1, false, false);
    }

    // ---- epilogue: +bias, LayerNorm over 256 channels per row, ReLU ----
    __shared__ float s_sum[16];
    __shared__ float s_sq[16];
    __shared__ float s_dp[16];
    if (tid < 16) { s_sum[tid] = 0.f; s_sq[tid] = 0.f; s_dp[tid] = 0.f; }
    __syncthreads();

    const float bc0 = bias[col0];
    const float bc1 = bias[col1];

    float f0[8], f1[8];
#pragma unroll
    for (int r = 0; r < 8; ++r) {
        f0[r] = c0[r] + bc0;
        f1[r] = c1[r] + bc1;
        float s = f0[r] + f1[r];
        float q = f0[r] * f0[r] + f1[r] * f1[r];
#pragma unroll
        for (int m = 8; m >= 1; m >>= 1) {
            s += __shfl_xor(s, m, 32);
            q += __shfl_xor(q, m, 32);
        }
        if (lo16 == 0) {
            atomicAdd(&s_sum[r + 8 * hi], s);   // ds_add_f32
            atomicAdd(&s_sq [r + 8 * hi], q);
        }
    }
    __syncthreads();

    const float g0  = gamma[col0], g1  = gamma[col1];
    const float be0 = beta [col0], be1 = beta [col1];
    float lw0 = 0.f, lw1 = 0.f;
    if (DUR) { lw0 = lin_w[col0]; lw1 = lin_w[col1]; }

#pragma unroll
    for (int r = 0; r < 8; ++r) {
        const int   row  = r + 8 * hi;            // ISA C-matrix row mapping
        const float mean = s_sum[row] * (1.0f / 256.0f);
        const float var  = s_sq[row] * (1.0f / 256.0f) - mean * mean;
        const float rstd = rsqrtf(var + 1e-5f);
        const float y0 = fmaxf((f0[r] - mean) * rstd * g0 + be0, 0.f);
        const float y1 = fmaxf((f1[r] - mean) * rstd * g1 + be1, 0.f);
        if (!DUR) {
            const size_t mrow = (size_t)blockIdx.x * 16 + row;
            Hout[mrow * DIM + col0] = (bf16_t)y0;
            Hout[mrow * DIM + col1] = (bf16_t)y1;
        } else {
            float dp = y0 * lw0 + y1 * lw1;
#pragma unroll
            for (int m = 8; m >= 1; m >>= 1) dp += __shfl_xor(dp, m, 32);
            if (lo16 == 0) atomicAdd(&s_dp[row], dp);
        }
    }

    if (DUR) {
        __syncthreads();
        if (tid < 16) {
            const float d = s_dp[tid] + lin_b[0];
            dur[(size_t)blockIdx.x * 16 + tid] = __expf(fmaxf(d, 0.f));
        }
    }
}

// ---------------- cumsum of target (16 rows of 512) ------------------------

__global__ void cumsum_kernel(const int* __restrict__ target, int* __restrict__ cum) {
    const int n = threadIdx.x;
    if (n < NB) {
        const int* t = target + n * LSEQ;
        int*       c = cum    + n * LSEQ;
        int acc = 0;
        for (int i = 0; i < LSEQ; ++i) { acc += t[i]; c[i] = acc; }
    }
}

// ---------------- searchsorted gather, streaming (nontemporal) stores ------
// block = 256 threads; each block emits 4 mel frames (4 x 256 floats).
// x (8MB) stays L2-resident; output (67MB) is written once -> nontemporal.

__global__ __launch_bounds__(256)
void gather_kernel(const float* __restrict__ x, const int* __restrict__ cum,
                   float* __restrict__ out)
{
    const int b = blockIdx.x;              // 0 .. 16*1024-1
    const int n = b >> 10;
    const int t = ((b & 1023) << 2) + (threadIdx.x >> 6);
    const int e = threadIdx.x & 63;        // float4 lane within the 256-ch row
    const int* c = cum + n * LSEQ;
    const int total = c[LSEQ - 1];

    f4 v = {0.f, 0.f, 0.f, 0.f};
    if (t < total) {
        int lo = 0, hiB = LSEQ;            // searchsorted 'right'
        while (lo < hiB) {
            int mid = (lo + hiB) >> 1;
            if (c[mid] <= t) lo = mid + 1; else hiB = mid;
        }
        int idx = (lo > LSEQ - 1) ? (LSEQ - 1) : lo;
        v = ((const f4*)(x + (size_t)(n * LSEQ + idx) * DIM))[e];
    }
    __builtin_nontemporal_store(v, (f4*)(out + (size_t)(n * TMAX + t) * DIM) + e);
}

// ---------------------------------------------------------------------------

extern "C" void kernel_launch(void* const* d_in, const int* in_sizes, int n_in,
                              void* d_out, int out_size, void* d_ws, size_t ws_size,
                              hipStream_t stream)
{
    const float* x      = (const float*)d_in[0];
    const int*   target = (const int*)  d_in[1];
    // d_in[2] = mel_max_length (fixed 4096)
    const float* w1  = (const float*)d_in[3];
    const float* b1  = (const float*)d_in[4];
    const float* g1  = (const float*)d_in[5];
    const float* be1 = (const float*)d_in[6];
    const float* w2  = (const float*)d_in[7];
    const float* b2  = (const float*)d_in[8];
    const float* g2  = (const float*)d_in[9];
    const float* be2 = (const float*)d_in[10];
    const float* lw  = (const float*)d_in[11];
    const float* lb  = (const float*)d_in[12];

    char* ws = (char*)d_ws;
    bf16_t* xb  = (bf16_t*)(ws);                 // 16*512*256 bf16  = 4,194,304 B
    bf16_t* w1b = (bf16_t*)(ws + 4194304);       // 256*768 bf16    =   393,216 B
    bf16_t* w2b = (bf16_t*)(ws + 4587520);       //                 =   393,216 B
    bf16_t* h1b = (bf16_t*)(ws + 4980736);       // 16*510*256 bf16 = 4,177,920 B
    int*    cum = (int*)   (ws + 9158656);       // 16*512 int      =    32,768 B

    float* out = (float*)d_out;                          // 16*4096*256
    float* dur = out + (size_t)NB * TMAX * DIM;          // 16*508

    cast_x_bf16 <<<(NB * LSEQ * DIM + 255) / 256, 256, 0, stream>>>(x, xb, NB * LSEQ * DIM);
    pack_w_bf16 <<<(256 * KRED + 255) / 256, 256, 0, stream>>>(w1, w1b);
    pack_w_bf16 <<<(256 * KRED + 255) / 256, 256, 0, stream>>>(w2, w2b);

    // conv1: M = 16*510 rows, Lin = 512
    conv_ln_wmma<false><<<510, 256, 0, stream>>>(xb,  w1b, b1, g1, be1, h1b,
                                                 nullptr, nullptr, nullptr, 510, 512);
    // conv2 + duration head: M = 16*508 rows, Lin = 510
    conv_ln_wmma<true> <<<508, 256, 0, stream>>>(h1b, w2b, b2, g2, be2, nullptr,
                                                 lw, lb, dur, 508, 510);

    cumsum_kernel<<<1, 32, 0, stream>>>(target, cum);
    gather_kernel<<<NB * (TMAX / 4), 256, 0, stream>>>(x, cum, out);
}
